// MultiHeadDeformableAttention_46351287058964
// MI455X (gfx1250) — matrix operand order repaired and checked by reading, not compiled, not run on hardware
//
#include <hip/hip_runtime.h>
#include <math.h>

#define BQ_TOTAL 16384   // B*Q = 4*4096
#define BATCH    4
#define CIN      256
#define COUT     256
#define NSAMP    8
#define HF       128
#define WF       128
#define HWSZ     (HF * WF)
#define KDIM     256     // every GEMM in this model reduces over 256

typedef __attribute__((ext_vector_type(16))) __bf16 v16bf;
typedef __attribute__((ext_vector_type(8)))  __bf16 v8bf;
typedef __attribute__((ext_vector_type(8)))  float  v8f;

// ---------------------------------------------------------------------------
// Elementwise f32 -> bf16 convert (count must be a multiple of 4)
// ---------------------------------------------------------------------------
__global__ __launch_bounds__(256) void convert_f32_bf16(
    const float* __restrict__ src, __bf16* __restrict__ dst) {
  const size_t i = ((size_t)blockIdx.x * blockDim.x + threadIdx.x) * 4;
  const float4 v = *(const float4*)(src + i);
  dst[i + 0] = (__bf16)v.x;
  dst[i + 1] = (__bf16)v.y;
  dst[i + 2] = (__bf16)v.z;
  dst[i + 3] = (__bf16)v.w;
}

// ---------------------------------------------------------------------------
// Tiled transpose with convert: src [rows, cols] f32 -> dst [cols, rows] bf16
// 32x32 tiles via LDS, block (32,8), batched via grid.z.
// ---------------------------------------------------------------------------
__global__ __launch_bounds__(256) void transpose_f32_bf16(
    const float* __restrict__ src, __bf16* __restrict__ dst,
    int rows, int cols, long long sstride, long long dstride) {
  __shared__ float tile[32][33];
  src += (size_t)blockIdx.z * sstride;
  dst += (size_t)blockIdx.z * dstride;

  const int x  = blockIdx.x * 32 + threadIdx.x;       // src col
  const int y0 = blockIdx.y * 32 + threadIdx.y;       // src row base
#pragma unroll
  for (int i = 0; i < 4; ++i) {
    const int y = y0 + i * 8;
    if (x < cols && y < rows)
      tile[threadIdx.y + i * 8][threadIdx.x] = src[(size_t)y * cols + x];
  }
  __syncthreads();
  const int tx  = blockIdx.y * 32 + threadIdx.x;      // dst col (= src row)
  const int ty0 = blockIdx.x * 32 + threadIdx.y;      // dst row (= src col) base
#pragma unroll
  for (int i = 0; i < 4; ++i) {
    const int ty = ty0 + i * 8;
    if (tx < rows && ty < cols)
      dst[(size_t)ty * rows + tx] = (__bf16)tile[threadIdx.x][threadIdx.y + i * 8];
  }
}

// ---------------------------------------------------------------------------
// WMMA GEMM, all-bf16 operands, K = 256 fixed, no bounds checks.
//   A  [M, K]  row-major bf16 (K contiguous)
//   Bt [N, K]  row-major bf16 (i.e. B transposed; K contiguous)
//   C  [M, N]  f32 or bf16, bias[n] added, optional tanh.
// Block = 256 threads = 8 waves arranged WM x WN (16x16 tile each).
// ---------------------------------------------------------------------------
template <int WM, int WN, bool OUT_BF16, bool ACT_TANH>
__global__ __launch_bounds__(256) void wmma_gemm(
    const __bf16* __restrict__ A, const __bf16* __restrict__ Bt,
    const float* __restrict__ bias, void* __restrict__ Cv,
    int N, long long astride, long long bstride, long long cstride) {
  A += (size_t)blockIdx.z * astride;
  Bt += (size_t)blockIdx.z * bstride;

  const int wave = threadIdx.x >> 5;
  const int lane = threadIdx.x & 31;
  const int wm = wave % WM;
  const int wn = wave / WM;
  const int row0 = (blockIdx.y * WM + wm) * 16;
  const int col0 = (blockIdx.x * WN + wn) * 16;

  // A operand: lane -> row, K run of 16 starting at kb
  const int am = row0 + (lane & 15);
  const int kb = (lane >> 4) << 4;     // 0 or 16
  // B operand: lane -> col, K run of 16 starting at kr
  const int bn = col0 + (lane & 15);
  const int kr = (lane >> 4) << 4;     // 0 or 16

  const __bf16* ap = A + (size_t)am * KDIM + kb;
  const __bf16* bp = Bt + (size_t)bn * KDIM + kr;

  v8f acc = {};
#pragma unroll
  for (int k0 = 0; k0 < KDIM; k0 += 32) {
    const v16bf a = *(const v16bf*)(ap + k0);
    const v16bf b = *(const v16bf*)(bp + k0);
    acc = __builtin_amdgcn_wmma_f32_16x16x32_bf16(
        false, a, false, b, (short)0, acc, false, false);
  }

  // C/D layout: element r -> row = row0 + (lane<16 ? r : 8+r), col = lane&15
  const int n = col0 + (lane & 15);
  const int mbase = (lane >> 4) << 3;
  const float bb = bias[n];
  if constexpr (OUT_BF16) {
    __bf16* C = (__bf16*)Cv + (size_t)blockIdx.z * cstride;
#pragma unroll
    for (int r = 0; r < 8; ++r) {
      float v = acc[r] + bb;
      if constexpr (ACT_TANH) v = tanhf(v);
      C[(size_t)(row0 + mbase + r) * N + n] = (__bf16)v;
    }
  } else {
    float* C = (float*)Cv + (size_t)blockIdx.z * cstride;
#pragma unroll
    for (int r = 0; r < 8; ++r) {
      float v = acc[r] + bb;
      if constexpr (ACT_TANH) v = tanhf(v);
      C[(size_t)(row0 + mbase + r) * N + n] = v;
    }
  }
}

// ---------------------------------------------------------------------------
// Deformable bilinear sampling + per-query attention over S=8 points.
// kpT/vpT are [B, H*W, C_OUT] bf16 -> all 256 channels of one spatial position
// are contiguous => every corner fetch is a coalesced 64B wave read.
// One block per query, wave w == head w (32 lanes == 32 head dims, wave32).
// ---------------------------------------------------------------------------
__global__ __launch_bounds__(256) void sample_attn_kernel(
    const __bf16* __restrict__ kpT, const __bf16* __restrict__ vpT,
    const float* __restrict__ loc, const __bf16* __restrict__ qh,
    __bf16* __restrict__ attn_out) {
  const int bq = blockIdx.x;
  const int b = bq >> 12;              // Q = 4096
  const int c = threadIdx.x;           // channel = head*32 + d
  const size_t mapbase = (size_t)b * HWSZ * COUT + c;
  const float qv = (float)qh[(size_t)bq * COUT + c];

  float logit[NSAMP];
  float vsamp[NSAMP];

#pragma unroll
  for (int s = 0; s < NSAMP; ++s) {
    const float lx = loc[(size_t)bq * (NSAMP * 2) + 2 * s + 0];
    const float ly = loc[(size_t)bq * (NSAMP * 2) + 2 * s + 1];
    // align_corners=False mapping
    const float gx = (lx + 1.f) * (WF * 0.5f) - 0.5f;
    const float gy = (ly + 1.f) * (HF * 0.5f) - 0.5f;
    const float x0f = floorf(gx), y0f = floorf(gy);
    const int x0 = (int)x0f, y0 = (int)y0f;
    const float wx = gx - x0f, wy = gy - y0f;
    const float w00 = (1.f - wx) * (1.f - wy);
    const float w01 = wx * (1.f - wy);
    const float w10 = (1.f - wx) * wy;
    const float w11 = wx * wy;
    const bool xa = (x0 >= 0) && (x0 < WF);
    const bool xb = (x0 + 1 >= 0) && (x0 + 1 < WF);
    const bool ya = (y0 >= 0) && (y0 < HF);
    const bool yb = (y0 + 1 >= 0) && (y0 + 1 < HF);
    const size_t p00 = mapbase + (size_t)(y0 * WF + x0) * COUT;

    const float k00 = (xa && ya) ? (float)kpT[p00] : 0.f;
    const float k01 = (xb && ya) ? (float)kpT[p00 + COUT] : 0.f;
    const float k10 = (xa && yb) ? (float)kpT[p00 + (size_t)WF * COUT] : 0.f;
    const float k11 = (xb && yb) ? (float)kpT[p00 + (size_t)(WF + 1) * COUT] : 0.f;
    const float v00 = (xa && ya) ? (float)vpT[p00] : 0.f;
    const float v01 = (xb && ya) ? (float)vpT[p00 + COUT] : 0.f;
    const float v10 = (xa && yb) ? (float)vpT[p00 + (size_t)WF * COUT] : 0.f;
    const float v11 = (xb && yb) ? (float)vpT[p00 + (size_t)(WF + 1) * COUT] : 0.f;

    const float ks = k00 * w00 + k01 * w01 + k10 * w10 + k11 * w11;
    vsamp[s] = v00 * w00 + v01 * w01 + v10 * w10 + v11 * w11;

    // per-head dot over d=32 lanes (wave32 butterfly reduction)
    float dot = qv * ks;
#pragma unroll
    for (int off = 16; off > 0; off >>= 1) dot += __shfl_xor(dot, off, 32);
    logit[s] = dot * 0.0625f;          // 1/sqrt(C_OUT) = 1/16
  }

  float mx = logit[0];
#pragma unroll
  for (int s = 1; s < NSAMP; ++s) mx = fmaxf(mx, logit[s]);
  float sum = 0.f;
#pragma unroll
  for (int s = 0; s < NSAMP; ++s) { logit[s] = __expf(logit[s] - mx); sum += logit[s]; }
  const float inv = 1.f / sum;

  float o = 0.f;
#pragma unroll
  for (int s = 0; s < NSAMP; ++s) o += vsamp[s] * logit[s];
  attn_out[(size_t)bq * COUT + c] = (__bf16)(o * inv);
}

// ---------------------------------------------------------------------------
static inline char* carve(char*& p, size_t bytes) {
  char* r = p;
  p += (bytes + 255) & ~(size_t)255;
  return r;
}

extern "C" void kernel_launch(void* const* d_in, const int* in_sizes, int n_in,
                              void* d_out, int out_size, void* d_ws, size_t ws_size,
                              hipStream_t stream) {
  (void)in_sizes; (void)n_in; (void)out_size; (void)ws_size;

  const float* queries = (const float*)d_in[0];
  const float* keys    = (const float*)d_in[1];
  const float* values  = (const float*)d_in[2];
  const float* W_loc   = (const float*)d_in[3];
  const float* b_loc   = (const float*)d_in[4];
  const float* W_q     = (const float*)d_in[5];
  const float* b_q     = (const float*)d_in[6];
  const float* W_k     = (const float*)d_in[7];
  const float* b_k     = (const float*)d_in[8];
  const float* W_v     = (const float*)d_in[9];
  const float* b_v     = (const float*)d_in[10];
  const float* W_o     = (const float*)d_in[11];
  const float* b_o     = (const float*)d_in[12];
  float* out = (float*)d_out;

  char* p = (char*)d_ws;
  __bf16* q_bf   = (__bf16*)carve(p, (size_t)BQ_TOTAL * CIN * 2);
  __bf16* wlocT  = (__bf16*)carve(p, (size_t)16 * CIN * 2);
  __bf16* wqT    = (__bf16*)carve(p, (size_t)COUT * CIN * 2);
  __bf16* woT    = (__bf16*)carve(p, (size_t)COUT * COUT * 2);
  __bf16* wk_bf  = (__bf16*)carve(p, (size_t)COUT * CIN * 2);
  __bf16* wv_bf  = (__bf16*)carve(p, (size_t)COUT * CIN * 2);
  __bf16* keysT  = (__bf16*)carve(p, (size_t)BATCH * HWSZ * CIN * 2);
  __bf16* valsT  = (__bf16*)carve(p, (size_t)BATCH * HWSZ * CIN * 2);
  float*  ws_loc = (float*)carve(p, (size_t)BQ_TOTAL * 16 * 4);
  __bf16* qh_bf  = (__bf16*)carve(p, (size_t)BQ_TOTAL * COUT * 2);
  __bf16* kpT    = (__bf16*)carve(p, (size_t)BATCH * HWSZ * COUT * 2);
  __bf16* vpT    = (__bf16*)carve(p, (size_t)BATCH * HWSZ * COUT * 2);
  __bf16* ao_bf  = (__bf16*)carve(p, (size_t)BQ_TOTAL * COUT * 2);

  const dim3 blk(256);
  const dim3 tblk(32, 8);
  const long long chw = (long long)CIN * HWSZ;   // per-batch elems of keys/keysT
  const long long ohw = (long long)COUT * HWSZ;  // per-batch elems of kpT

  // ---- pre-pass: bf16 conversion + K-contiguous layouts -------------------
  convert_f32_bf16<<<dim3((BQ_TOTAL * CIN) / 1024), blk, 0, stream>>>(queries, q_bf);
  convert_f32_bf16<<<dim3((COUT * CIN) / 1024), blk, 0, stream>>>(W_k, wk_bf);
  convert_f32_bf16<<<dim3((COUT * CIN) / 1024), blk, 0, stream>>>(W_v, wv_bf);
  transpose_f32_bf16<<<dim3(1, CIN / 32, 1), tblk, 0, stream>>>(W_loc, wlocT, CIN, 16, 0, 0);
  transpose_f32_bf16<<<dim3(COUT / 32, CIN / 32, 1), tblk, 0, stream>>>(W_q, wqT, CIN, COUT, 0, 0);
  transpose_f32_bf16<<<dim3(COUT / 32, COUT / 32, 1), tblk, 0, stream>>>(W_o, woT, COUT, COUT, 0, 0);
  transpose_f32_bf16<<<dim3(HWSZ / 32, CIN / 32, BATCH), tblk, 0, stream>>>(keys, keysT, CIN, HWSZ, chw, chw);
  transpose_f32_bf16<<<dim3(HWSZ / 32, CIN / 32, BATCH), tblk, 0, stream>>>(values, valsT, CIN, HWSZ, chw, chw);

  // ---- loc = tanh(queries @ W_loc + b_loc)  [16384 x 16] f32 --------------
  wmma_gemm<8, 1, false, true><<<dim3(1, BQ_TOTAL / 128, 1), blk, 0, stream>>>(
      q_bf, wlocT, b_loc, ws_loc, 16, 0, 0, 0);

  // ---- qh = queries @ W_q + b_q             [16384 x 256] bf16 ------------
  wmma_gemm<2, 4, true, false><<<dim3(COUT / 64, BQ_TOTAL / 32, 1), blk, 0, stream>>>(
      q_bf, wqT, b_q, qh_bf, COUT, 0, 0, 0);

  // ---- k_projT[b] = (W_k @ keys[b])^T + b_k  [hw x 256] bf16 --------------
  wmma_gemm<2, 4, true, false><<<dim3(COUT / 64, HWSZ / 32, BATCH), blk, 0, stream>>>(
      keysT, wk_bf, b_k, kpT, COUT, chw, 0, ohw);

  // ---- v_projT ------------------------------------------------------------
  wmma_gemm<2, 4, true, false><<<dim3(COUT / 64, HWSZ / 32, BATCH), blk, 0, stream>>>(
      valsT, wv_bf, b_v, vpT, COUT, chw, 0, ohw);

  // ---- deformable sampling + attention (L2-resident bf16 maps) ------------
  sample_attn_kernel<<<dim3(BQ_TOTAL), blk, 0, stream>>>(kpT, vpT, ws_loc, qh_bf, ao_bf);

  // ---- out = attn_out @ W_o + b_o           [16384 x 256] f32 -------------
  wmma_gemm<2, 4, false, false><<<dim3(COUT / 64, BQ_TOTAL / 32, 1), blk, 0, stream>>>(
      ao_bf, woT, b_o, out, COUT, 0, 0, 0);
}
